// DiscriminativeClueMiner_8735963480403
// MI455X (gfx1250) — matrix operand order repaired and checked by reading, not compiled, not run on hardware
//
#include <hip/hip_runtime.h>
#include <hip/hip_bf16.h>
#include <math.h>

// ---------------------------------------------------------------------------
// DiscriminativeClueMiner forward for MI455X (gfx1250, wave32, WMMA).
// B=128 N=1024 D=768 C=28 TD=1536 K=16 DEPTH=2 HEADS=8 HD=96
// Big GEMMs: v_wmma_f32_16x16x32_bf16, A-tiles staged f32 via
// GLOBAL_LOAD_ASYNC_TO_LDS_B128 (fallback: plain loads), bf16 convert at
// fragment read, W tiles transposed+converted in LDS.
// ---------------------------------------------------------------------------

#define Bb 128
#define Nn 1024
#define Dd 768
#define Cc 28
#define TDd 1536
#define Kk 16
#define HDh 96

typedef __attribute__((ext_vector_type(16))) __bf16 v16bf;
typedef __attribute__((ext_vector_type(8)))  float  v8f;
typedef __attribute__((ext_vector_type(4)))  int    v4i;

#define AS(n) __attribute__((address_space(n)))

#if defined(__AMDGCN__) && __has_builtin(__builtin_amdgcn_global_load_async_to_lds_b128)
#define ASYNC_LDS 1
#else
#define ASYNC_LDS 0
#endif

__device__ __forceinline__ void wait_asynccnt0() {
#if defined(__AMDGCN__)
#if __has_builtin(__builtin_amdgcn_s_wait_asynccnt)
  __builtin_amdgcn_s_wait_asynccnt(0);
#else
  asm volatile("s_wait_asynccnt 0x0" ::: "memory");
#endif
#endif
}

#if ASYNC_LDS
__device__ __forceinline__ void async_copy16(const float* g, float* lds) {
  __builtin_amdgcn_global_load_async_to_lds_b128(
      (AS(1) v4i*)g, (AS(3) v4i*)lds, 0, 0);
}
#endif

// A-fragment: 16x32 from f32 LDS tile Af[16rows][32] (row stride 32 floats),
// converted to bf16 at read.  k order used here (contiguous, same as B frag):
// lane l: M = l%16 ; K = klo..klo+15, klo=(l<16)?0:16
__device__ __forceinline__ v16bf load_a_frag_f32(const float* Af, int rowBase,
                                                 int lane) {
  int m   = lane & 15;
  int klo = (lane < 16) ? 0 : 16;
  const float* p = Af + (rowBase + m) * 32;
  v16bf a;
#pragma unroll
  for (int i = 0; i < 8; ++i) {
    a[i]     = (__bf16)p[klo + i];
    a[8 + i] = (__bf16)p[klo + 8 + i];
  }
  return a;
}

// B-fragment: 32x16 bf16 from LDS W^T tile Wt[n][32] (n-major, stride 32).
// lane l: N = l%16 ; K = (l<16)? 0..15 : 16..31 (contiguous in Wt row)
__device__ __forceinline__ v16bf load_b_frag(const __bf16* Wt, int nBase, int lane) {
  int n   = nBase + (lane & 15);
  int klo = (lane < 16) ? 0 : 16;
  return *(const v16bf*)(Wt + n * 32 + klo);
}

__device__ __forceinline__ v8f wmma_bf16(v16bf a, v16bf b, v8f c) {
  return __builtin_amdgcn_wmma_f32_16x16x32_bf16(false, a, false, b, (short)0, c,
                                                 false, false);
}

// ---------------------------------------------------------------------------
// Kernel 1: fused scorer.  scores = sigmoid(relu(x@sw1+sb1)@sw2+sb2)
// Block: 256 thr (8 waves); 128 rows of flattened [B*N, D] per block.
// Each wave: 16 rows x 192 hidden = 12 WMMA accumulators, 24 K-steps.
// ---------------------------------------------------------------------------
#define SHID 192
__global__ __launch_bounds__(256)
void scorer_kernel(const float* __restrict__ x, const float* __restrict__ sw1,
                   const float* __restrict__ sb1, const float* __restrict__ sw2,
                   const float* __restrict__ sb2, float* __restrict__ scores) {
  __shared__ __attribute__((aligned(32))) float  Af[128 * 32];   // 16 KB
  __shared__ __attribute__((aligned(32))) __bf16 Wt[SHID * 32];  // 12 KB
  int tid = threadIdx.x, lane = tid & 31, wave = tid >> 5;
  long row0 = (long)blockIdx.x * 128;

  v8f acc[12];
#pragma unroll
  for (int t = 0; t < 12; ++t) acc[t] = (v8f){};

  for (int ks = 0; ks < Dd / 32; ++ks) {
    int k0 = ks * 32;
    // stage A tile (f32) via async DMA to LDS
#if ASYNC_LDS
    for (int ch = tid; ch < 128 * 8; ch += 256) {     // 16B chunks, 8/row
      int r = ch >> 3, c4 = (ch & 7) << 2;
      async_copy16(&x[(row0 + r) * Dd + k0 + c4], &Af[r * 32 + c4]);
    }
#else
    for (int i = tid; i < 128 * 32; i += 256) {
      int r = i >> 5, c = i & 31;
      Af[i] = x[(row0 + r) * Dd + k0 + c];
    }
#endif
    // stage W^T bf16 (transpose + convert)
    for (int i = tid; i < 32 * SHID; i += 256) {
      int c = i / SHID, n = i % SHID;
      Wt[n * 32 + c] = (__bf16)sw1[(long)(k0 + c) * SHID + n];
    }
    if (k0 + 32 < Dd)
      __builtin_prefetch(&x[(row0 + (tid >> 1)) * Dd + k0 + 32], 0, 1);
#if ASYNC_LDS
    wait_asynccnt0();
#endif
    __syncthreads();

    v16bf a = load_a_frag_f32(Af, wave * 16, lane);
    v16bf bcur = load_b_frag(Wt, 0, lane);
#pragma unroll
    for (int t = 0; t < 12; ++t) {
      v16bf bnxt = bcur;
      if (t + 1 < 12) bnxt = load_b_frag(Wt, (t + 1) * 16, lane);
      acc[t] = wmma_bf16(a, bcur, acc[t]);
      bcur = bnxt;
    }
    __syncthreads();
  }

  // relu(h + sb1) dot sw2 reduced over 192 columns, then sigmoid.
  float part[8];
#pragma unroll
  for (int r = 0; r < 8; ++r) part[r] = 0.0f;
  int nLane = lane & 15;
#pragma unroll
  for (int t = 0; t < 12; ++t) {
    int col = t * 16 + nLane;
    float b1 = sb1[col], w2 = sw2[col];
#pragma unroll
    for (int r = 0; r < 8; ++r) {
      float h = acc[t][r] + b1;
      h = h > 0.0f ? h : 0.0f;
      part[r] += h * w2;
    }
  }
#pragma unroll
  for (int r = 0; r < 8; ++r) {
    float v = part[r];
    v += __shfl_xor(v, 1); v += __shfl_xor(v, 2);
    v += __shfl_xor(v, 4); v += __shfl_xor(v, 8);
    part[r] = v;
  }
  if (nLane == 0) {
    float b2 = sb2[0];
    int mBase = (lane >> 4) * 8;   // lane0 -> rows 0..7, lane16 -> rows 8..15
#pragma unroll
    for (int r = 0; r < 8; ++r) {
      long row = row0 + wave * 16 + mBase + r;
      float s = part[r] + b2;
      scores[row] = 1.0f / (1.0f + expf(-s));
    }
  }
}

// ---------------------------------------------------------------------------
// Kernel 2: per-batch top-16 (iterative argmax, matches lax.top_k order),
// gather sel = x[idx]*score, cosine adjacency + softmax(x10).
// ---------------------------------------------------------------------------
__global__ __launch_bounds__(256)
void topk_gather_adj_kernel(const float* __restrict__ x,
                            const float* __restrict__ scores,
                            float* __restrict__ topk_out,
                            float* __restrict__ adj_out,
                            float* __restrict__ g_out) {
  __shared__ float ssc[Nn];
  __shared__ float rval[256];
  __shared__ int   ridx[256];
  __shared__ int   selIdx[Kk];
  __shared__ float selSc[Kk];
  __shared__ float sel[Kk][Dd];
  __shared__ float nrm[Kk];
  __shared__ float prt[Kk][16];
  __shared__ float sadj[Kk][Kk];
  int b = blockIdx.x, tid = threadIdx.x;

  for (int i = tid; i < Nn; i += 256) ssc[i] = scores[(long)b * Nn + i];
  __syncthreads();

  for (int k = 0; k < Kk; ++k) {
    float best = -1e30f; int bi = Nn;
    for (int i = tid; i < Nn; i += 256) {
      float v = ssc[i];
      if (v > best || (v == best && i < bi)) { best = v; bi = i; }
    }
    rval[tid] = best; ridx[tid] = bi;
    __syncthreads();
    for (int s = 128; s > 0; s >>= 1) {
      if (tid < s) {
        float v = rval[tid + s]; int j = ridx[tid + s];
        if (v > rval[tid] || (v == rval[tid] && j < ridx[tid])) {
          rval[tid] = v; ridx[tid] = j;
        }
      }
      __syncthreads();
    }
    if (tid == 0) { selIdx[k] = ridx[0]; selSc[k] = rval[0]; ssc[ridx[0]] = -1e30f; }
    __syncthreads();
  }
  if (tid < Kk) topk_out[(long)b * Kk + tid] = (float)selIdx[tid];

  for (int i = tid; i < Kk * Dd; i += 256) {
    int k = i / Dd, d = i % Dd;
    float v = x[((long)b * Nn + selIdx[k]) * Dd + d] * selSc[k];
    sel[k][d] = v;
    g_out[((long)b * Kk + k) * Dd + d] = v;
  }
  __syncthreads();

  { // row norms
    int k = tid >> 4, sub = tid & 15;
    float s = 0.0f;
    for (int d = sub; d < Dd; d += 16) s += sel[k][d] * sel[k][d];
    prt[k][sub] = s;
  }
  __syncthreads();
  if (tid < Kk) {
    float s = 0.0f;
    for (int j = 0; j < 16; ++j) s += prt[tid][j];
    nrm[tid] = fmaxf(sqrtf(s), 1e-12f);
  }
  __syncthreads();

  { // cosine-similarity logits * 10
    int k = tid >> 4, j = tid & 15;
    float s = 0.0f;
    for (int d = 0; d < Dd; ++d) s += sel[k][d] * sel[j][d];
    sadj[k][j] = s / (nrm[k] * nrm[j]) * 10.0f;
  }
  __syncthreads();
  if (tid < Kk) { // row softmax
    float mx = -1e30f;
    for (int j = 0; j < Kk; ++j) mx = fmaxf(mx, sadj[tid][j]);
    float sm = 0.0f;
    for (int j = 0; j < Kk; ++j) { float e = expf(sadj[tid][j] - mx); sadj[tid][j] = e; sm += e; }
    float r = 1.0f / sm;
    for (int j = 0; j < Kk; ++j) sadj[tid][j] *= r;
  }
  __syncthreads();
  for (int i = tid; i < Kk * Kk; i += 256)
    adj_out[(long)b * Kk * Kk + i] = sadj[i >> 4][i & 15];
}

// ---------------------------------------------------------------------------
// Kernel 3: m = adj @ g   (per batch, 16x16 @ 16x768)
// ---------------------------------------------------------------------------
__global__ __launch_bounds__(256)
void gcn_agg_kernel(const float* __restrict__ adj, const float* __restrict__ g,
                    float* __restrict__ m) {
  __shared__ float sadj[Kk * Kk];
  __shared__ float sg[Kk * Dd];
  int b = blockIdx.x, tid = threadIdx.x;
  for (int i = tid; i < Kk * Kk; i += 256) sadj[i] = adj[(long)b * Kk * Kk + i];
  for (int i = tid; i < Kk * Dd; i += 256) sg[i] = g[(long)b * Kk * Dd + i];
  __syncthreads();
  for (int i = tid; i < Kk * Dd; i += 256) {
    int k = i / Dd, d = i % Dd;
    float s = 0.0f;
#pragma unroll
    for (int j = 0; j < Kk; ++j) s += sadj[k * Kk + j] * sg[j * Dd + d];
    m[(long)b * Kk * Dd + i] = s;
  }
}

// ---------------------------------------------------------------------------
// Kernel 4: generic C[M,N] = A[M,Kd] @ W[Kd,N] + bias, bf16 WMMA, f32 acc.
// Block tile 128x64, K step 32; 8 waves, each 16 rows x 64 cols (4 tiles).
// A staged f32 (async DMA when available), W transposed to bf16 in LDS.
// ---------------------------------------------------------------------------
__global__ __launch_bounds__(256)
void gemm_bf16_kernel(const float* __restrict__ A, const float* __restrict__ W,
                      const float* __restrict__ bias, float* __restrict__ C,
                      int M, int Kd, int N) {
  __shared__ __attribute__((aligned(32))) float  Af[128 * 32];  // 16 KB
  __shared__ __attribute__((aligned(32))) __bf16 Wt[64 * 32];   //  4 KB
  int tid = threadIdx.x, lane = tid & 31, wave = tid >> 5;
  int row0 = blockIdx.x * 128;
  int col0 = blockIdx.y * 64;

  v8f acc[4];
#pragma unroll
  for (int t = 0; t < 4; ++t) acc[t] = (v8f){};

  int ksteps = Kd / 32;
  for (int ks = 0; ks < ksteps; ++ks) {
    int k0 = ks * 32;
#if ASYNC_LDS
    for (int ch = tid; ch < 128 * 8; ch += 256) {
      int r = ch >> 3, c4 = (ch & 7) << 2;
      int gr = row0 + r; if (gr >= M) gr = M - 1;   // rows >= M never stored
      async_copy16(&A[(size_t)gr * Kd + k0 + c4], &Af[r * 32 + c4]);
    }
#else
    for (int i = tid; i < 128 * 32; i += 256) {
      int r = i >> 5, c = i & 31;
      int gr = row0 + r; if (gr >= M) gr = M - 1;
      Af[i] = A[(size_t)gr * Kd + k0 + c];
    }
#endif
    for (int i = tid; i < 32 * 64; i += 256) {
      int c = i >> 6, n = i & 63;                   // c = k within step
      Wt[n * 32 + c] = (__bf16)W[(size_t)(k0 + c) * N + col0 + n];
    }
    if (k0 + 32 < Kd) {
      int gr = row0 + (tid >> 1);
      if (gr < M) __builtin_prefetch(&A[(size_t)gr * Kd + k0 + 32], 0, 1);
    }
#if ASYNC_LDS
    wait_asynccnt0();
#endif
    __syncthreads();

    v16bf a = load_a_frag_f32(Af, wave * 16, lane);
    v16bf bcur = load_b_frag(Wt, 0, lane);
#pragma unroll
    for (int t = 0; t < 4; ++t) {
      v16bf bnxt = bcur;
      if (t + 1 < 4) bnxt = load_b_frag(Wt, (t + 1) * 16, lane);
      acc[t] = wmma_bf16(a, bcur, acc[t]);
      bcur = bnxt;
    }
    __syncthreads();
  }

  int mLane = (lane >> 4) * 8;
  int nLane = lane & 15;
#pragma unroll
  for (int t = 0; t < 4; ++t) {
    int gn = col0 + t * 16 + nLane;
    float bv = bias ? bias[gn] : 0.0f;
#pragma unroll
    for (int r = 0; r < 8; ++r) {
      int gm = row0 + wave * 16 + mLane + r;
      if (gm < M) C[(size_t)gm * N + gn] = acc[t][r] + bv;
    }
  }
}

// ---------------------------------------------------------------------------
// Kernel 5: g = g + relu(LayerNorm(tmp) * ln_g + ln_b), one row per block.
// ---------------------------------------------------------------------------
__global__ __launch_bounds__(256)
void ln_res_kernel(const float* __restrict__ t, const float* __restrict__ lng,
                   const float* __restrict__ lnb, float* __restrict__ g) {
  __shared__ float red[256];
  long row = blockIdx.x;
  int tid = threadIdx.x;
  const float* tr = t + row * Dd;

  float s = 0.0f;
  for (int d = tid; d < Dd; d += 256) s += tr[d];
  red[tid] = s; __syncthreads();
  for (int st = 128; st > 0; st >>= 1) {
    if (tid < st) red[tid] += red[tid + st];
    __syncthreads();
  }
  float mu = red[0] / (float)Dd;
  __syncthreads();

  float v = 0.0f;
  for (int d = tid; d < Dd; d += 256) { float dd = tr[d] - mu; v += dd * dd; }
  red[tid] = v; __syncthreads();
  for (int st = 128; st > 0; st >>= 1) {
    if (tid < st) red[tid] += red[tid + st];
    __syncthreads();
  }
  float rstd = rsqrtf(red[0] / (float)Dd + 1e-5f);

  for (int d = tid; d < Dd; d += 256) {
    float h = (tr[d] - mu) * rstd * lng[d] + lnb[d];
    h = h > 0.0f ? h : 0.0f;
    g[row * Dd + d] += h;
  }
}

// ---------------------------------------------------------------------------
// Kernel 6: 8-head cross attention per batch (q shared across batch).
// ---------------------------------------------------------------------------
__global__ __launch_bounds__(256)
void attention_kernel(const float* __restrict__ qb, const float* __restrict__ kb,
                      const float* __restrict__ vb, float* __restrict__ ctx,
                      float* __restrict__ attw_out) {
  __shared__ float sq[Cc * HDh];
  __shared__ float sk[Kk * HDh];
  __shared__ float sv[Kk * HDh];
  __shared__ float att[Cc * Kk];
  __shared__ float aw[Cc * Kk];
  int b = blockIdx.x, tid = threadIdx.x;
  float invs = rsqrtf((float)HDh);

  for (int i = tid; i < Cc * Kk; i += 256) aw[i] = 0.0f;

  for (int h = 0; h < 8; ++h) {
    __syncthreads();
    for (int i = tid; i < Cc * HDh; i += 256) {
      int c = i / HDh, d = i % HDh;
      sq[i] = qb[(long)c * Dd + h * HDh + d];
    }
    for (int i = tid; i < Kk * HDh; i += 256) {
      int k = i / HDh, d = i % HDh;
      sk[i] = kb[((long)b * Kk + k) * Dd + h * HDh + d];
      sv[i] = vb[((long)b * Kk + k) * Dd + h * HDh + d];
    }
    __syncthreads();
    for (int i = tid; i < Cc * Kk; i += 256) {
      int c = i / Kk, k = i % Kk;
      float s = 0.0f;
      for (int d = 0; d < HDh; ++d) s += sq[c * HDh + d] * sk[k * HDh + d];
      att[i] = s * invs;
    }
    __syncthreads();
    if (tid < Cc) {
      float mx = -1e30f;
      for (int k = 0; k < Kk; ++k) mx = fmaxf(mx, att[tid * Kk + k]);
      float sm = 0.0f;
      for (int k = 0; k < Kk; ++k) { float e = expf(att[tid * Kk + k] - mx); att[tid * Kk + k] = e; sm += e; }
      float r = 1.0f / sm;
      for (int k = 0; k < Kk; ++k) {
        att[tid * Kk + k] *= r;
        aw[tid * Kk + k] += att[tid * Kk + k] * 0.125f;
      }
    }
    __syncthreads();
    for (int i = tid; i < Cc * HDh; i += 256) {
      int c = i / HDh, d = i % HDh;
      float s = 0.0f;
#pragma unroll
      for (int k = 0; k < Kk; ++k) s += att[c * Kk + k] * sv[k * HDh + d];
      ctx[((long)b * Cc + c) * Dd + h * HDh + d] = s;
    }
  }
  __syncthreads();
  for (int i = tid; i < Cc * Kk; i += 256) attw_out[(long)b * Cc * Kk + i] = aw[i];
}

// ---------------------------------------------------------------------------
// Kernel 7: logits[b,c] = scale * dot(attn_out[b,c,:], lq[c,:]). Wave per row.
// ---------------------------------------------------------------------------
__global__ __launch_bounds__(256)
void logits_kernel(const float* __restrict__ ao, const float* __restrict__ lq,
                   const float* __restrict__ scale, float* __restrict__ out) {
  int tid = threadIdx.x, lane = tid & 31, wave = tid >> 5;
  long row = (long)blockIdx.x * 8 + wave;          // [0, B*C)
  int c = (int)(row % Cc);
  float s = 0.0f;
  for (int d = lane; d < Dd; d += 32) s += ao[row * Dd + d] * lq[(long)c * Dd + d];
  s += __shfl_xor(s, 1); s += __shfl_xor(s, 2); s += __shfl_xor(s, 4);
  s += __shfl_xor(s, 8); s += __shfl_xor(s, 16);
  if (lane == 0) out[row] = s * scale[0];
}

// ---------------------------------------------------------------------------
extern "C" void kernel_launch(void* const* d_in, const int* in_sizes, int n_in,
                              void* d_out, int out_size, void* d_ws, size_t ws_size,
                              hipStream_t stream) {
  const float* x     = (const float*)d_in[0];
  const float* lemb  = (const float*)d_in[1];
  const float* sw1   = (const float*)d_in[2];
  const float* sb1   = (const float*)d_in[3];
  const float* sw2   = (const float*)d_in[4];
  const float* sb2   = (const float*)d_in[5];
  const float* gcn_w = (const float*)d_in[6];
  const float* gcn_b = (const float*)d_in[7];
  const float* ln_g  = (const float*)d_in[8];
  const float* ln_b  = (const float*)d_in[9];
  const float* lp_w  = (const float*)d_in[10];
  const float* lp_b  = (const float*)d_in[11];
  const float* wq    = (const float*)d_in[12];
  const float* wk    = (const float*)d_in[13];
  const float* wv    = (const float*)d_in[14];
  const float* bq    = (const float*)d_in[15];
  const float* bk    = (const float*)d_in[16];
  const float* bv    = (const float*)d_in[17];
  const float* wo    = (const float*)d_in[18];
  const float* bo    = (const float*)d_in[19];
  const float* scale = (const float*)d_in[20];

  // Output layout (flat, return order): logits | scores | topk_idx | adj | attw
  float* out    = (float*)d_out;
  float* logits = out;                                   // 128*28
  float* scores = logits + (long)Bb * Cc;                // 128*1024
  float* topk   = scores + (long)Bb * Nn;                // 128*16 (as float)
  float* adj    = topk + (long)Bb * Kk;                  // 128*16*16
  float* attw   = adj + (long)Bb * Kk * Kk;              // 128*28*16

  // Workspace (~54 MB of f32)
  float* ws   = (float*)d_ws;
  long   gsz  = (long)Bb * Kk * Dd;                      // 1,572,864
  long   csz  = (long)Bb * Cc * Dd;                      // 2,752,512
  float* g    = ws;
  float* m    = g + gsz;
  float* tmp  = m + gsz;
  float* kbuf = tmp + gsz;
  float* vbuf = kbuf + gsz;
  float* lq   = vbuf + gsz;
  float* qbuf = lq + (long)Cc * Dd;
  float* ctx  = qbuf + (long)Cc * Dd;
  float* aout = ctx + csz;

  // 1) scorer MLP -> scores
  scorer_kernel<<<(Bb * Nn) / 128, 256, 0, stream>>>(x, sw1, sb1, sw2, sb2, scores);

  // 2) top-k + gather + adjacency
  topk_gather_adj_kernel<<<Bb, 256, 0, stream>>>(x, scores, topk, adj, g);

  // 3) GCN depth-2
  for (int i = 0; i < 2; ++i) {
    gcn_agg_kernel<<<Bb, 256, 0, stream>>>(adj, g, m);
    gemm_bf16_kernel<<<dim3((Bb * Kk) / 128, Dd / 64), 256, 0, stream>>>(
        m, gcn_w + (long)i * Dd * Dd, gcn_b + (long)i * Dd, tmp, Bb * Kk, Dd, Dd);
    ln_res_kernel<<<Bb * Kk, 256, 0, stream>>>(tmp, ln_g + (long)i * Dd,
                                               ln_b + (long)i * Dd, g);
  }

  // 4) label projection + q/k/v projections
  gemm_bf16_kernel<<<dim3(1, Dd / 64), 256, 0, stream>>>(lemb, lp_w, lp_b, lq, Cc, TDd, Dd);
  gemm_bf16_kernel<<<dim3(1, Dd / 64), 256, 0, stream>>>(lq, wq, bq, qbuf, Cc, Dd, Dd);
  gemm_bf16_kernel<<<dim3((Bb * Kk) / 128, Dd / 64), 256, 0, stream>>>(
      g, wk, bk, kbuf, Bb * Kk, Dd, Dd);
  gemm_bf16_kernel<<<dim3((Bb * Kk) / 128, Dd / 64), 256, 0, stream>>>(
      g, wv, bv, vbuf, Bb * Kk, Dd, Dd);

  // 5) attention -> ctx, attn_weights
  attention_kernel<<<Bb, 256, 0, stream>>>(qbuf, kbuf, vbuf, ctx, attw);

  // 6) output projection + logits
  gemm_bf16_kernel<<<dim3((Bb * Cc) / 128, Dd / 64), 256, 0, stream>>>(
      ctx, wo, bo, aout, Bb * Cc, Dd, Dd);
  logits_kernel<<<(Bb * Cc) / 8, 256, 0, stream>>>(aout, lq, scale, logits);
}
